// HierarchicalGraphEncoder_22368189677715
// MI455X (gfx1250) — compile-verified, matrix operand order checked by reading, not run on hardware
//
#include <hip/hip_runtime.h>
#include <hip/hip_bf16.h>
#include <math.h>

// ---------------- problem dims ----------------
#define BTn  2048     // B*T
#define Dm   256
#define Hn   4
#define HDm  64
#define NFf  75       // fine nodes
#define NHh  4        // high nodes
#define Lnum 2

#define ACT_NONE 0
#define ACT_GELU 1
#define ACT_SIG  2

typedef __attribute__((ext_vector_type(16))) _Float16 v16h;
typedef __attribute__((ext_vector_type(8)))  _Float16 v8h;
typedef __attribute__((ext_vector_type(4)))  _Float16 v4h;
typedef __attribute__((ext_vector_type(8)))  float    v8f;

// ---------------- wave32 helpers ----------------
__device__ __forceinline__ float warp_sum(float v){
#pragma unroll
  for (int o = 16; o; o >>= 1) v += __shfl_xor(v, o, 32);
  return v;
}
__device__ __forceinline__ float warp_max(float v){
#pragma unroll
  for (int o = 16; o; o >>= 1) v = fmaxf(v, __shfl_xor(v, o, 32));
  return v;
}
__device__ __forceinline__ float gelu_f(float x){
  return 0.5f * x * (1.f + erff(x * 0.70710678118654752f));
}
__device__ __forceinline__ float sigmoid_f(float x){
  return 1.f / (1.f + __expf(-x));
}
__device__ __forceinline__ v8f wmma32(v16h a, v16h b, v8f c){
  return __builtin_amdgcn_wmma_f32_16x16x32_f16(false, a, false, b, (short)0, c, false, false);
}

// -------- unguarded fragment loads (interior tiles only) --------
// p = row base + lane k-offset; 16B aligned. 4x global_load_b128.
__device__ __forceinline__ v16h ldMK_full(const float* __restrict__ p){
  const float4 a0 = *(const float4*)(p + 0);
  const float4 a1 = *(const float4*)(p + 4);
  const float4 b0 = *(const float4*)(p + 16);
  const float4 b1 = *(const float4*)(p + 20);
  v16h f;
  f[0]=(_Float16)a0.x;  f[1]=(_Float16)a0.y;  f[2]=(_Float16)a0.z;  f[3]=(_Float16)a0.w;
  f[4]=(_Float16)a1.x;  f[5]=(_Float16)a1.y;  f[6]=(_Float16)a1.z;  f[7]=(_Float16)a1.w;
  f[8]=(_Float16)b0.x;  f[9]=(_Float16)b0.y;  f[10]=(_Float16)b0.z; f[11]=(_Float16)b0.w;
  f[12]=(_Float16)b1.x; f[13]=(_Float16)b1.y; f[14]=(_Float16)b1.z; f[15]=(_Float16)b1.w;
  return f;
}
// p = column base (+ lane k-half * ld); strided by ld, no guards.
__device__ __forceinline__ v16h ldKN_full(const float* __restrict__ p, int ld){
  v16h f;
#pragma unroll
  for (int e = 0; e < 16; ++e){
    int off = e + ((e & 8) ? 8 : 0);
    f[e] = (_Float16)p[(long)off * ld];
  }
  return f;
}

// -------- guarded generic loaders (tails / attention) --------
__device__ __forceinline__ v16h ldfrag_MK(const float* __restrict__ Mp, int ld,
                                          int r0, int k0, int nrows, int ncols, int ln){
  v16h f;
  int r   = r0 + (ln & 15);
  int kb  = k0 + ((ln & 16) ? 8 : 0);
  bool rok = (r < nrows);
  const float* p = Mp + (long)(rok ? r : 0) * ld + kb;
  if (rok && (kb + 24 <= ncols) && ((((size_t)p) & 15) == 0)){
    f = ldMK_full(p);
  } else {
    const float* row = Mp + (long)(rok ? r : 0) * ld;
#pragma unroll
    for (int e = 0; e < 16; ++e){
      int k  = kb + e + ((e & 8) ? 8 : 0);
      bool ok = rok && (k < ncols);
      float v = row[ok ? k : 0];               // clamped address, branchless
      f[e] = ok ? (_Float16)v : (_Float16)0.f;
    }
  }
  return f;
}
__device__ __forceinline__ v16h ldfrag_KN(const float* __restrict__ Mp, int ld,
                                          int k0, int n0, int nk, int ncols, int ln){
  v16h f;
  int n  = n0 + (ln & 15);
  int kb = k0 + ((ln & 16) ? 8 : 0);
  bool nok = (n < ncols);
  const float* base = Mp + (nok ? n : 0);
#pragma unroll
  for (int e = 0; e < 16; ++e){
    int k  = kb + e + ((e & 8) ? 8 : 0);
    bool ok = nok && (k < nk);
    float v = base[(long)(ok ? k : 0) * ld];   // clamped address, branchless
    f[e] = ok ? (_Float16)v : (_Float16)0.f;
  }
  return f;
}

// ---------------- generic batched WMMA GEMM ----------------
// Out[z] = act(alpha * A[z] @ Bop + bias) (+ Res[z])
// Bop: bKN==0 -> Wm is NxK row-major (A@W^T); bKN==1 -> Wm is KxN row-major.
// Block = 4 waves; block tile = 16 x 256; each wave owns a 16x64 strip (4 acc).
// grid: x = ceil(N/256), y = ceil(M/16), z = frames.  (all N here are multiples of 256)
// A tile (16x32) staged through LDS as f16 once per block per k-step; the A
// fragment is read with 2x ds_load_b128 and reused by 4 back-to-back v_wmma.
// The bKN branch is hoisted: two specialized k-loops so the register allocator
// can pin accumulators without per-iteration phi copies. A single guarded step
// handles K % 32 tails.
__global__ void __launch_bounds__(128)
gemm_wmma_k(const float* __restrict__ A, long sA, int lda,
            const float* __restrict__ Wm, long sW, int ldw, int bKN,
            const float* __restrict__ bias,
            const float* __restrict__ Res, long sR, int ldr,
            float* __restrict__ Out, long sO, int ldo,
            int M, int N, int K, float alpha, int act)
{
  __shared__ _Float16 Atile[16 * 32];
  int z   = blockIdx.z;
  int tid = threadIdx.x;
  int wv  = tid >> 5, ln = tid & 31;
  int lrow  = ln & 15;
  int lhalf = (ln & 16) ? 8 : 0;
  int nbase = blockIdx.x * 256 + wv * 64;
  int m0    = blockIdx.y * 16;
  const float* Af = A  + (long)z * sA;
  const float* Wf = Wm + (long)z * sW;

  // cooperative A-tile fill (16x32 floats -> f16 in LDS)
  auto fillA = [&](int k0){
    int idx = tid * 4;
    int rr = idx >> 5, cc = idx & 31;
    int gr = m0 + rr;
    bool grok = (gr < M);
    const float* rowp = Af + (long)(grok ? gr : 0) * lda;
    v4h hq;
    if (grok && (k0 + cc + 4 <= K) && ((((size_t)(rowp + k0 + cc)) & 15) == 0)){
      float4 v = *(const float4*)(rowp + k0 + cc);   // global_load_b128
      hq[0] = (_Float16)v.x; hq[1] = (_Float16)v.y;
      hq[2] = (_Float16)v.z; hq[3] = (_Float16)v.w;
    } else {
#pragma unroll
      for (int u = 0; u < 4; ++u){
        bool ok = grok && (k0 + cc + u < K);
        float v = rowp[ok ? (k0 + cc + u) : 0];
        hq[u] = ok ? (_Float16)v : (_Float16)0.f;
      }
    }
    *(v4h*)&Atile[rr * 32 + cc] = hq;                // ds_store_b64
  };
  auto ldsA = [&](){
    v16h a;
    v8h lo = *(const v8h*)&Atile[lrow * 32 + lhalf];        // ds_load_b128
    v8h hi = *(const v8h*)&Atile[lrow * 32 + lhalf + 16];   // ds_load_b128
#pragma unroll
    for (int e = 0; e < 8; ++e){ a[e] = lo[e]; a[8 + e] = hi[e]; }
    return a;
  };

  v8f acc0 = {}, acc1 = {}, acc2 = {}, acc3 = {};
  int Kmain = K & ~31;

  if (bKN){
    const float* cb = Wf + nbase + lrow + (long)lhalf * ldw;
    const float* bp0 = cb;
    const float* bp1 = cb + 16;
    const float* bp2 = cb + 32;
    const float* bp3 = cb + 48;
    long bstep = (long)32 * ldw;
    for (int k0 = 0; k0 < Kmain; k0 += 32){
      __syncthreads();
      fillA(k0);
      __syncthreads();
      v16h a = ldsA();
      acc0 = wmma32(a, ldKN_full(bp0, ldw), acc0);
      acc1 = wmma32(a, ldKN_full(bp1, ldw), acc1);
      acc2 = wmma32(a, ldKN_full(bp2, ldw), acc2);
      acc3 = wmma32(a, ldKN_full(bp3, ldw), acc3);
      bp0 += bstep; bp1 += bstep; bp2 += bstep; bp3 += bstep;
    }
    if (Kmain < K){
      __syncthreads();
      fillA(Kmain);
      __syncthreads();
      v16h a = ldsA();
      acc0 = wmma32(a, ldfrag_KN(Wf, ldw, Kmain, nbase +  0, K, N, ln), acc0);
      acc1 = wmma32(a, ldfrag_KN(Wf, ldw, Kmain, nbase + 16, K, N, ln), acc1);
      acc2 = wmma32(a, ldfrag_KN(Wf, ldw, Kmain, nbase + 32, K, N, ln), acc2);
      acc3 = wmma32(a, ldfrag_KN(Wf, ldw, Kmain, nbase + 48, K, N, ln), acc3);
    }
  } else {
    const float* bp0 = Wf + (long)(nbase +  0 + lrow) * ldw + lhalf;
    const float* bp1 = Wf + (long)(nbase + 16 + lrow) * ldw + lhalf;
    const float* bp2 = Wf + (long)(nbase + 32 + lrow) * ldw + lhalf;
    const float* bp3 = Wf + (long)(nbase + 48 + lrow) * ldw + lhalf;
    for (int k0 = 0; k0 < Kmain; k0 += 32){
      __syncthreads();
      fillA(k0);
      __syncthreads();
      v16h a = ldsA();
      acc0 = wmma32(a, ldMK_full(bp0), acc0);
      acc1 = wmma32(a, ldMK_full(bp1), acc1);
      acc2 = wmma32(a, ldMK_full(bp2), acc2);
      acc3 = wmma32(a, ldMK_full(bp3), acc3);
      bp0 += 32; bp1 += 32; bp2 += 32; bp3 += 32;
    }
    if (Kmain < K){
      __syncthreads();
      fillA(Kmain);
      __syncthreads();
      v16h a = ldsA();
      acc0 = wmma32(a, ldfrag_MK(Wf, ldw, nbase +  0, Kmain, N, K, ln), acc0);
      acc1 = wmma32(a, ldfrag_MK(Wf, ldw, nbase + 16, Kmain, N, K, ln), acc1);
      acc2 = wmma32(a, ldfrag_MK(Wf, ldw, nbase + 32, Kmain, N, K, ln), acc2);
      acc3 = wmma32(a, ldfrag_MK(Wf, ldw, nbase + 48, Kmain, N, K, ln), acc3);
    }
  }

  int mb = m0 + ((ln & 16) ? 8 : 0);
  float* Of = Out + (long)z * sO;
  const float* Rf = Res ? (Res + (long)z * sR) : nullptr;
  v8f* accp[4] = {&acc0, &acc1, &acc2, &acc3};
#pragma unroll
  for (int j = 0; j < 4; ++j){
    int n = nbase + j * 16 + lrow;
    if (n < N){
      float bv = bias ? bias[n] : 0.f;
      v8f av = *accp[j];
#pragma unroll
      for (int c = 0; c < 8; ++c){
        int m = mb + c;
        if (m < M){
          float v = av[c] * alpha + bv;
          if (act == ACT_GELU)      v = gelu_f(v);
          else if (act == ACT_SIG)  v = sigmoid_f(v);
          if (Rf) v += Rf[(long)m * ldr + n];
          Of[(long)m * ldo + n] = v;
        }
      }
    }
  }
}

// ---------------- per-head Q@K^T scores (WMMA) ----------------
__global__ void __launch_bounds__(128)
attn_scores_k(const float* __restrict__ Q, long sQ, int ldq, int qoff,
              const float* __restrict__ Km, long sK, int ldk, int koff,
              float* __restrict__ S, int Nn, float alpha)
{
  int bt = blockIdx.z, h = blockIdx.y;
  int wv = threadIdx.x >> 5, ln = threadIdx.x & 31;
  int nt = (Nn + 15) >> 4;
  int tiles = nt * nt;
  const float* Qf = Q  + (long)bt * sQ + qoff + h * HDm;
  const float* Kf = Km + (long)bt * sK + koff + h * HDm;
  float* Sf = S + ((long)bt * Hn + h) * (long)Nn * Nn;
  for (int t = blockIdx.x * 4 + wv; t < tiles; t += gridDim.x * 4){
    int mt = t / nt, nc = t % nt;
    v8f acc = {};
#pragma unroll
    for (int k0 = 0; k0 < HDm; k0 += 32){
      v16h a = ldfrag_MK(Qf, ldq, mt * 16, k0, Nn, HDm, ln);
      v16h b = ldfrag_MK(Kf, ldk, nc * 16, k0, Nn, HDm, ln);
      acc = wmma32(a, b, acc);
    }
    int mcol = nc * 16 + (ln & 15);
    int rb   = mt * 16 + ((ln & 16) ? 8 : 0);
    if (mcol < Nn){
#pragma unroll
      for (int c = 0; c < 8; ++c){
        int r = rb + c;
        if (r < Nn) Sf[(long)r * Nn + mcol] = acc[c] * alpha;
      }
    }
  }
}

// ---------------- attention apply: O[:, h*64+d] = S_h @ V_h ----------------
__global__ void __launch_bounds__(128)
attn_apply_k(const float* __restrict__ S,
             const float* __restrict__ V, long sV, int ldv, int voff,
             float* __restrict__ O, long sO, int ldo, int Nn)
{
  int bt = blockIdx.z, h = blockIdx.y;
  int wv = threadIdx.x >> 5, ln = threadIdx.x & 31;
  int m0 = blockIdx.x * 16;
  int n0 = wv * 16;                              // within HD=64
  const float* Sf = S + ((long)bt * Hn + h) * (long)Nn * Nn;
  const float* Vf = V + (long)bt * sV + voff + h * HDm;
  v8f acc = {};
  for (int k0 = 0; k0 < Nn; k0 += 32){
    v16h a = ldfrag_MK(Sf, Nn, m0, k0, Nn, Nn, ln);
    v16h b = ldfrag_KN(Vf, ldv, k0, n0, Nn, HDm, ln);
    acc = wmma32(a, b, acc);
  }
  float* Ofb = O + (long)bt * sO;
  int col = h * HDm + n0 + (ln & 15);
  int rb  = m0 + ((ln & 16) ? 8 : 0);
#pragma unroll
  for (int c = 0; c < 8; ++c){
    int m = rb + c;
    if (m < Nn) Ofb[(long)m * ldo + col] = acc[c];
  }
}

// ---------------- softmax(+pos) over m, mean over heads -> Am ----------------
__global__ void softmax_mean_k(const float* __restrict__ S, const float* __restrict__ pos,
                               float* __restrict__ Am, int Nn, int rows)
{
  int row = blockIdx.x * 4 + (threadIdx.x >> 5);
  if (row >= rows) return;
  int ln = threadIdx.x & 31;
  int bt = row / Nn, n = row % Nn;
  float accv[3] = {0.f, 0.f, 0.f};
  for (int h = 0; h < Hn; ++h){
    const float* Sr = S + (((long)bt * Hn + h) * Nn + n) * (long)Nn;
    float tmp[3];
    float mx = -3.0e38f;
#pragma unroll
    for (int j = 0; j < 3; ++j){ int m = ln + j * 32;
      if (m < Nn){ float v = Sr[m] + pos[n * Nn + m]; tmp[j] = v; mx = fmaxf(mx, v); } }
    mx = warp_max(mx);
    float sm = 0.f;
#pragma unroll
    for (int j = 0; j < 3; ++j){ int m = ln + j * 32;
      if (m < Nn){ float e = __expf(tmp[j] - mx); tmp[j] = e; sm += e; } }
    sm = warp_sum(sm);
    float inv = 0.25f / sm;   // includes mean over 4 heads
#pragma unroll
    for (int j = 0; j < 3; ++j){ int m = ln + j * 32; if (m < Nn) accv[j] += tmp[j] * inv; }
  }
  float* Ar = Am + ((long)bt * Nn + n) * (long)Nn;
#pragma unroll
  for (int j = 0; j < 3; ++j){ int m = ln + j * 32; if (m < Nn) Ar[m] = accv[j]; }
}

// ---------------- in-place row softmax ----------------
__global__ void softmax_rows_k(float* __restrict__ S, int Nn, int rows)
{
  int row = blockIdx.x * 4 + (threadIdx.x >> 5);
  if (row >= rows) return;
  int ln = threadIdx.x & 31;
  float* Sr = S + (long)row * Nn;
  float tmp[3];
  float mx = -3.0e38f;
#pragma unroll
  for (int j = 0; j < 3; ++j){ int m = ln + j * 32;
    if (m < Nn){ tmp[j] = Sr[m]; mx = fmaxf(mx, tmp[j]); } }
  mx = warp_max(mx);
  float sm = 0.f;
#pragma unroll
  for (int j = 0; j < 3; ++j){ int m = ln + j * 32;
    if (m < Nn){ tmp[j] = __expf(tmp[j] - mx); sm += tmp[j]; } }
  sm = warp_sum(sm);
  float inv = 1.f / sm;
#pragma unroll
  for (int j = 0; j < 3; ++j){ int m = ln + j * 32; if (m < Nn) Sr[m] = tmp[j] * inv; }
}

// ---------------- LayerNorm over D=256, one wave per row ----------------
__global__ void ln_rows_k(const float* __restrict__ X, int ldx,
                          float* __restrict__ O, int ldo,
                          const float* __restrict__ g, const float* __restrict__ b,
                          int nrows, int act)
{
  int row = blockIdx.x * 4 + (threadIdx.x >> 5);
  if (row >= nrows) return;
  int ln = threadIdx.x & 31;
  const float* xr = X + (long)row * ldx;
  float v[8]; float s = 0.f;
#pragma unroll
  for (int j = 0; j < 8; ++j){ v[j] = xr[ln + j * 32]; s += v[j]; }
  s = warp_sum(s);
  float mean = s * (1.f / 256.f);
  float var = 0.f;
#pragma unroll
  for (int j = 0; j < 8; ++j){ float d = v[j] - mean; var += d * d; }
  var = warp_sum(var) * (1.f / 256.f);
  float inv = rsqrtf(var + 1e-5f);
  float* orow = O + (long)row * ldo;
#pragma unroll
  for (int j = 0; j < 8; ++j){
    int c = ln + j * 32;
    float o = (v[j] - mean) * inv * g[c] + b[c];
    if (act == ACT_GELU) o = gelu_f(o);
    orow[c] = o;
  }
}

// ---------------- small elementwise / pooling kernels ----------------
__global__ void concat_fine_k(const float* __restrict__ xp, const float* __restrict__ xl,
                              const float* __restrict__ xr, float* __restrict__ fine)
{
  long i = (long)blockIdx.x * blockDim.x + threadIdx.x;
  long total = (long)BTn * NFf * Dm;
  if (i >= total) return;
  int c = (int)(i % Dm); long t = i / Dm;
  int node = (int)(t % NFf); long bt = t / NFf;
  float v;
  if (node < 33)      v = xp[(bt * 33 + node) * Dm + c];
  else if (node < 54) v = xl[(bt * 21 + (node - 33)) * Dm + c];
  else                v = xr[(bt * 21 + (node - 54)) * Dm + c];
  fine[i] = v;
}

__global__ void gate_combine_k(const float* __restrict__ h, const float* __restrict__ g,
                               const float* __restrict__ ctx, float* __restrict__ out, long total)
{
  long i = (long)blockIdx.x * blockDim.x + threadIdx.x;
  if (i >= total) return;
  out[i] = h[i] + g[i] * ctx[i];
}

__global__ void pool_k(const float* __restrict__ fine, const float* __restrict__ face,
                       float* __restrict__ pooled)
{
  int bt = blockIdx.x, c = threadIdx.x;
  const float* f = fine + (long)bt * NFf * Dm + c;
  float s0 = 0.f, s1 = 0.f, s2 = 0.f;
  for (int n = 0;  n < 33; ++n) s0 += f[n * Dm];
  for (int n = 33; n < 54; ++n) s1 += f[n * Dm];
  for (int n = 54; n < 75; ++n) s2 += f[n * Dm];
  float* p = pooled + (long)bt * 4 * Dm;
  p[c]          = s0 * (1.f / 33.f);
  p[Dm + c]     = s1 * (1.f / 21.f);
  p[2 * Dm + c] = s2 * (1.f / 21.f);
  p[3 * Dm + c] = face[(long)bt * Dm + c];
}

__global__ void fuse_k(const float* __restrict__ fine, const float* __restrict__ high,
                       const float* __restrict__ face, float* __restrict__ fused)
{
  int bt = blockIdx.x, c = threadIdx.x;
  const float* f = fine + (long)bt * NFf * Dm + c;
  float s = 0.f;
  for (int n = 0; n < NFf; ++n) s += f[n * Dm];
  s *= (1.f / 75.f);
  const float* hh = high + (long)bt * NHh * Dm + c;
  float s2 = 0.f;
  for (int r = 0; r < NHh; ++r) s2 += hh[r * Dm];
  s2 *= 0.25f;
  fused[(long)bt * Dm + c] = s + s2 + face[(long)bt * Dm + c];
}

// ---------------- input layout (setup_inputs insertion order, depth-first) ----
enum InIdx {
  IN_X_POSE = 0, IN_X_LHAND, IN_X_RHAND, IN_X_RAW,
  FA_QW, FA_QB, FA_KW, FA_KB, FA_POS,                 // fine_adj
  FG_LNG, FG_LNB, FG_W, FG_B,                         // fine_gcn
  FT_LN1G, FT_LN1B, FT_LN2G, FT_LN2B, FT_INW, FT_INB, FT_OUTW, FT_OUTB,
  FT_GATEW, FT_GATEB, FT_W1, FT_B1, FT_W2, FT_B2,     // fine_tr
  HA_QW, HA_QB, HA_KW, HA_KB, HA_POS,                 // high_adj
  HG_LNG, HG_LNB, HG_W, HG_B,                         // high_gcn
  HT_LN1G, HT_LN1B, HT_LN2G, HT_LN2B, HT_INW, HT_INB, HT_OUTW, HT_OUTB,
  HT_GATEW, HT_GATEB, HT_W1, HT_B1, HT_W2, HT_B2,     // high_tr
  FC_W, FC_B, FC_LNG, FC_LNB,                         // face
  PL_W, PL_B, PL_LNG, PL_LNB,                         // pool
  OU_W, OU_B, OU_LNG, OU_LNB                          // out
};

extern "C" void kernel_launch(void* const* d_in, const int* in_sizes, int n_in,
                              void* d_out, int out_size, void* d_ws, size_t ws_size,
                              hipStream_t stream)
{
  auto F = [&](int i) -> const float* { return (const float*)d_in[i]; };

  // ---- workspace carve-up (f32) ----
  float* ws = (float*)d_ws;
  size_t off = 0;
  auto alloc = [&](size_t n){ float* p = ws + off; off += n; return p; };
  float* fine   = alloc((size_t)BTn * NFf * Dm);
  float* bufA   = alloc((size_t)BTn * NFf * Dm);
  float* bufB   = alloc((size_t)BTn * NFf * Dm);
  float* bufC   = alloc((size_t)BTn * NFf * Dm);
  float* qkv    = alloc((size_t)BTn * NFf * 1024);
  float* Sb     = alloc((size_t)BTn * Hn * NFf * NFf);
  float* Amf    = alloc((size_t)BTn * NFf * NFf);
  float* Ahm    = alloc((size_t)BTn * NHh * NHh);
  float* faceb  = alloc((size_t)BTn * Dm);
  float* pooled = alloc((size_t)BTn * 4 * Dm);
  float* high   = alloc((size_t)BTn * NHh * Dm);
  float* fused  = alloc((size_t)BTn * Dm);
  float* obuf   = alloc((size_t)BTn * Dm);

  auto gemm = [&](const float* A, long sA, int lda,
                  const float* Wm, long sW, int ldw, int bKN,
                  const float* bias,
                  const float* Res, long sR, int ldr,
                  float* Out, long sO, int ldo,
                  int M, int N, int K, int frames, float alpha, int act){
    dim3 g((N + 255) / 256, (M + 15) / 16, frames);
    gemm_wmma_k<<<g, 128, 0, stream>>>(A, sA, lda, Wm, sW, ldw, bKN, bias,
                                       Res, sR, ldr, Out, sO, ldo, M, N, K, alpha, act);
  };
  auto lnorm = [&](const float* X, int ldx, float* O, int ldo,
                   const float* g, const float* b, int rows, int act){
    ln_rows_k<<<dim3((rows + 3) / 4), 128, 0, stream>>>(X, ldx, O, ldo, g, b, rows, act);
  };

  // 1) build fine = concat(pose, lhand, rhand)
  {
    long tot = (long)BTn * NFf * Dm;
    concat_fine_k<<<dim3((unsigned)((tot + 255) / 256)), 256, 0, stream>>>(
        F(IN_X_POSE), F(IN_X_LHAND), F(IN_X_RHAND), fine);
  }

  // 2) fine adjacency: Q,K projections -> scores -> softmax(+pos), mean heads
  gemm(fine, (long)NFf * Dm, Dm, F(FA_QW), 0, Dm, 0, F(FA_QB),
       nullptr, 0, 0, bufA, (long)NFf * Dm, Dm, NFf, Dm, Dm, BTn, 1.f, ACT_NONE);
  gemm(fine, (long)NFf * Dm, Dm, F(FA_KW), 0, Dm, 0, F(FA_KB),
       nullptr, 0, 0, bufB, (long)NFf * Dm, Dm, NFf, Dm, Dm, BTn, 1.f, ACT_NONE);
  {
    int nt = (NFf + 15) / 16;
    dim3 g((nt * nt + 3) / 4, Hn, BTn);
    attn_scores_k<<<g, 128, 0, stream>>>(bufA, (long)NFf * Dm, Dm, 0,
                                         bufB, (long)NFf * Dm, Dm, 0, Sb, NFf, 0.125f);
  }
  softmax_mean_k<<<dim3((BTn * NFf + 3) / 4), 128, 0, stream>>>(Sb, F(FA_POS), Amf, NFf, BTn * NFf);

  // ---- shared GCN+Transformer layer stack ----
  auto run_layers = [&](float* X, int Nn, const float* Amat, int gB, int tB){
    long sX  = (long)Nn * Dm;
    long sQ  = (long)Nn * 768;
    long sFF = (long)Nn * 1024;
    long sAm = (long)Nn * Nn;
    for (int i = 0; i < Lnum; ++i){
      // GCN: x += gelu(lin(Am @ LN(x)))
      lnorm(X, Dm, bufA, Dm, F(gB + 0) + i * Dm, F(gB + 1) + i * Dm, BTn * Nn, ACT_NONE);
      gemm(Amat, sAm, Nn, bufA, sX, Dm, 1, nullptr,
           nullptr, 0, 0, bufB, sX, Dm, Nn, Dm, Nn, BTn, 1.f, ACT_NONE);
      gemm(bufB, sX, Dm, F(gB + 2) + (size_t)i * Dm * Dm, 0, Dm, 0, F(gB + 3) + i * Dm,
           X, sX, Dm, X, sX, Dm, Nn, Dm, Dm, BTn, 1.f, ACT_GELU);
      // TR: gated context + MHA + FFN
      lnorm(X, Dm, bufA, Dm, F(tB + 0) + i * Dm, F(tB + 1) + i * Dm, BTn * Nn, ACT_NONE);
      gemm(Amat, sAm, Nn, bufA, sX, Dm, 1, nullptr,
           nullptr, 0, 0, bufB, sX, Dm, Nn, Dm, Nn, BTn, 1.f, ACT_NONE);       // ctx
      gemm(bufB, sX, Dm, F(tB + 8) + (size_t)i * Dm * Dm, 0, Dm, 0, F(tB + 9) + i * Dm,
           nullptr, 0, 0, bufC, sX, Dm, Nn, Dm, Dm, BTn, 1.f, ACT_SIG);        // gate
      {
        long tot = (long)BTn * Nn * Dm;
        gate_combine_k<<<dim3((unsigned)((tot + 255) / 256)), 256, 0, stream>>>(
            bufA, bufC, bufB, bufC, tot);                                      // hg
      }
      gemm(bufC, sX, Dm, F(tB + 4) + (size_t)i * 768 * Dm, 0, Dm, 0, F(tB + 5) + i * 768,
           nullptr, 0, 0, qkv, sQ, 768, Nn, 768, Dm, BTn, 1.f, ACT_NONE);      // qkv
      {
        int nt = (Nn + 15) / 16;
        dim3 g((nt * nt + 3) / 4, Hn, BTn);
        attn_scores_k<<<g, 128, 0, stream>>>(qkv, sQ, 768, 0, qkv, sQ, 768, 256,
                                             Sb, Nn, 0.125f);
      }
      softmax_rows_k<<<dim3((BTn * Hn * Nn + 3) / 4), 128, 0, stream>>>(Sb, Nn, BTn * Hn * Nn);
      {
        dim3 g((Nn + 15) / 16, Hn, BTn);
        attn_apply_k<<<g, 128, 0, stream>>>(Sb, qkv, sQ, 768, 512, bufB, sX, Dm, Nn);
      }
      gemm(bufB, sX, Dm, F(tB + 6) + (size_t)i * Dm * Dm, 0, Dm, 0, F(tB + 7) + i * Dm,
           X, sX, Dm, X, sX, Dm, Nn, Dm, Dm, BTn, 1.f, ACT_NONE);              // out-proj + res
      lnorm(X, Dm, bufA, Dm, F(tB + 2) + i * Dm, F(tB + 3) + i * Dm, BTn * Nn, ACT_NONE);
      gemm(bufA, sX, Dm, F(tB + 10) + (size_t)i * 1024 * Dm, 0, Dm, 0, F(tB + 11) + i * 1024,
           nullptr, 0, 0, qkv, sFF, 1024, Nn, 1024, Dm, BTn, 1.f, ACT_GELU);   // ffn1
      gemm(qkv, sFF, 1024, F(tB + 12) + (size_t)i * Dm * 1024, 0, 1024, 0, F(tB + 13) + i * Dm,
           X, sX, Dm, X, sX, Dm, Nn, Dm, 1024, BTn, 1.f, ACT_NONE);            // ffn2 + res
    }
  };

  run_layers(fine, NFf, Amf, FG_LNG, FT_LN1G);

  // 3) face: rows = frames (M=2048), x_raw[:, :, 75:543, :] is contiguous 1872 per frame
  gemm(F(IN_X_RAW) + 300, 0, 543 * 4, F(FC_W), 0, 1872, 0, F(FC_B),
       nullptr, 0, 0, faceb, 0, Dm, BTn, Dm, 1872, 1, 1.f, ACT_NONE);
  lnorm(faceb, Dm, faceb, Dm, F(FC_LNG), F(FC_LNB), BTn, ACT_GELU);

  // 4) pooled regions -> high nodes
  pool_k<<<dim3(BTn), 256, 0, stream>>>(fine, faceb, pooled);
  for (int r = 0; r < 4; ++r){
    gemm(pooled + r * Dm, 0, 4 * Dm, F(PL_W) + (size_t)r * Dm * Dm, 0, Dm, 0, F(PL_B) + r * Dm,
         nullptr, 0, 0, high + r * Dm, 0, NHh * Dm, BTn, Dm, Dm, 1, 1.f, ACT_NONE);
    lnorm(high + r * Dm, NHh * Dm, high + r * Dm, NHh * Dm,
          F(PL_LNG) + r * Dm, F(PL_LNB) + r * Dm, BTn, ACT_NONE);
  }

  // 5) high adjacency
  gemm(high, (long)NHh * Dm, Dm, F(HA_QW), 0, Dm, 0, F(HA_QB),
       nullptr, 0, 0, bufA, (long)NHh * Dm, Dm, NHh, Dm, Dm, BTn, 1.f, ACT_NONE);
  gemm(high, (long)NHh * Dm, Dm, F(HA_KW), 0, Dm, 0, F(HA_KB),
       nullptr, 0, 0, bufB, (long)NHh * Dm, Dm, NHh, Dm, Dm, BTn, 1.f, ACT_NONE);
  {
    dim3 g(1, Hn, BTn);
    attn_scores_k<<<g, 128, 0, stream>>>(bufA, (long)NHh * Dm, Dm, 0,
                                         bufB, (long)NHh * Dm, Dm, 0, Sb, NHh, 0.125f);
  }
  softmax_mean_k<<<dim3((BTn * NHh + 3) / 4), 128, 0, stream>>>(Sb, F(HA_POS), Ahm, NHh, BTn * NHh);

  run_layers(high, NHh, Ahm, HG_LNG, HT_LN1G);

  // 6) fuse + output head
  fuse_k<<<dim3(BTn), 256, 0, stream>>>(fine, high, faceb, fused);
  gemm(fused, 0, Dm, F(OU_W), 0, Dm, 0, F(OU_B),
       nullptr, 0, 0, obuf, 0, Dm, BTn, Dm, Dm, 1, 1.f, ACT_NONE);
  lnorm(obuf, Dm, (float*)d_out, Dm, F(OU_LNG), F(OU_LNB), BTn, ACT_NONE);
}
